// ChunkSampler_32993938768368
// MI455X (gfx1250) — matrix operand order repaired and checked by reading, not compiled, not run on hardware
//
#include <hip/hip_runtime.h>
#include <hip/hip_bf16.h>

typedef __attribute__((ext_vector_type(2)))  float        v2f;
typedef __attribute__((ext_vector_type(4)))  float        v4f;
typedef __attribute__((ext_vector_type(8)))  float        v8f;
typedef __attribute__((ext_vector_type(8)))  __bf16       v8bf;
typedef __attribute__((ext_vector_type(16))) __bf16       v16bf;
typedef __attribute__((ext_vector_type(4)))  unsigned int uint4_t;
typedef __attribute__((ext_vector_type(8)))  int          int8_t_v;
typedef __attribute__((ext_vector_type(4)))  int          int4_t_v;

#define VOCAB 50257
#define DIM   1024
#define BATCH 64

#define CHUNK_K   64                 // floats of K staged per TDM op
#define ROW_PITCH 68                 // 64+4 pad DWORDs: rows stay 16B-aligned (272B=17*16)
#define BUF_DW    (16 * ROW_PITCH)   // one buffer: 16 rows * 68 dw = 4352 B
#define NWAVE     4                  // waves per block, one strip each

// ---------------------------------------------------------------------------
// Kernel 0: split H (64x1024 f32) into bf16 hi/lo planes (RNE split:
// hi = bf16_rne(x), lo = bf16(x - hi)), stored in WMMA-fragment order so the
// GEMM loads each A fragment as one contiguous v16bf (2 clause'd b128s).
// ---------------------------------------------------------------------------
__global__ __launch_bounds__(256) void split_hidden(
    const float* __restrict__ H, __bf16* __restrict__ Ahi, __bf16* __restrict__ Alo)
{
  const int i = blockIdx.x * 256 + threadIdx.x;
  if (i < BATCH * DIM) {
    const int r   = i >> 10;           // row
    const int K   = i & (DIM - 1);     // k index
    const int rem = K & 31;
    const int h   = rem >> 4;          // lane half
    const int run = (rem >> 3) & 1;    // K run 0/1
    const int j   = (rem & 7) + run * 8;
    const int dst = r * DIM + (K & ~31) + h * 16 + j;

    const float f = H[i];
    const __bf16 hb = (__bf16)f;       // RNE
    Ahi[dst] = hb;
    Alo[dst] = (__bf16)(f - (float)hb);
  }
}

// split 16 contiguous f32 (from LDS) into bf16 hi/lo fragments (RNE split:
// both hi and lo assemble via v_cvt_pk_bf16_f32 into contiguous VGPRs)
__device__ __forceinline__ void split16(const float* __restrict__ bp,
                                        v16bf& hi, v16bf& lo)
{
  v4f q0 = *(const v4f*)(bp + 0);
  v4f q1 = *(const v4f*)(bp + 4);
  v4f q2 = *(const v4f*)(bp + 8);
  v4f q3 = *(const v4f*)(bp + 12);
  union { __bf16 b[16]; v16bf v; } Hh, Ll;
#pragma unroll
  for (int i = 0; i < 16; ++i) {
    const float f = (i < 4 ? q0[i] : i < 8 ? q1[i - 4] : i < 12 ? q2[i - 8] : q3[i - 12]);
    const __bf16 h = (__bf16)f;        // RNE -> packed cvt pairs
    Hh.b[i] = h;
    Ll.b[i] = (__bf16)(f - (float)h);
  }
  hi = Hh.v; lo = Ll.v;
}

// ---------------------------------------------------------------------------
// Kernel 1: logits[b,v] = (H[b,:] . E[v,:]) / temp[b]
// One wave owns one 16-col strip of V, computing all 64 rows as four 16x16
// C tiles. Embedding strip streamed to LDS by the Tensor Data Mover
// (TENSORcnt double-buffered); GEMM is bf16x3 split-precision
// V_WMMA_F32_16X16X32_BF16 (Ahi*Bhi + Alo*Bhi + Ahi*Blo, f32 accumulate).
// ---------------------------------------------------------------------------
__global__ __launch_bounds__(128) void lmhead_gemm_wmma(
    const __bf16* __restrict__ Ahi, const __bf16* __restrict__ Alo,
    const float* __restrict__ E, const float* __restrict__ temps,
    float* __restrict__ logits)
{
  __shared__ float lds[NWAVE * 2 * BUF_DW];      // 34816 B

  const int lane = threadIdx.x & 31;
  const int wave = threadIdx.x >> 5;
  const int half = lane >> 4;
  const int ln   = lane & 15;

  const long strip = (long)blockIdx.x * NWAVE + wave;
  const long row0  = strip * 16;
  if (row0 >= VOCAB) return;                     // wave-uniform tail guard
  int rows = VOCAB - (int)row0; if (rows > 16) rows = 16;
  const long col = row0 + ln;

  float* mybuf = lds + (size_t)wave * 2 * BUF_DW;
  const unsigned lds_base = (unsigned)(size_t)mybuf;
  const unsigned long long gbase =
      (unsigned long long)(size_t)E + (unsigned long long)row0 * DIM * 4ull;

  auto tdm_issue = [&](int c) {
    const unsigned lds_addr = lds_base + (unsigned)(c & 1) * (BUF_DW * 4);
    const unsigned long long ga = gbase + (unsigned long long)c * (CHUNK_K * 4);
    uint4_t g0 = { 1u, lds_addr, (unsigned)ga,
                   ((unsigned)(ga >> 32) & 0x01FFFFFFu) | 0x80000000u };
    int8_t_v g1 = { (int)0x07520000,                       // 4B elems, pad 4dw/64dw
                    (int)((DIM & 0xFFFF) << 16),           // tensor_dim0=1024
                    (int)((DIM >> 16) | ((rows & 0xFFFF) << 16)),  // tensor_dim1=rows
                    (int)((rows >> 16) | (CHUNK_K << 16)), // tile_dim0=64
                    (int)16,                               // tile_dim1=16
                    (int)DIM,                              // dim0 stride
                    0, 0 };
    int4_t_v gz4 = { 0, 0, 0, 0 };
    int8_t_v gz8 = { 0, 0, 0, 0, 0, 0, 0, 0 };
    __builtin_amdgcn_tensor_load_to_lds(g0, g1, gz4, gz4, gz8, 0);
  };

  // fragment-ordered bf16 A-plane bases for the 4 M tiles
  const __bf16* Ah0 = Ahi + (size_t)( 0 + ln) * DIM + half * 16;
  const __bf16* Ah1 = Ahi + (size_t)(16 + ln) * DIM + half * 16;
  const __bf16* Ah2 = Ahi + (size_t)(32 + ln) * DIM + half * 16;
  const __bf16* Ah3 = Ahi + (size_t)(48 + ln) * DIM + half * 16;
  const __bf16* Al0 = Alo + (size_t)( 0 + ln) * DIM + half * 16;
  const __bf16* Al1 = Alo + (size_t)(16 + ln) * DIM + half * 16;
  const __bf16* Al2 = Alo + (size_t)(32 + ln) * DIM + half * 16;
  const __bf16* Al3 = Alo + (size_t)(48 + ln) * DIM + half * 16;

  v8f c0 = {}, c1 = {}, c2 = {}, c3 = {};

  tdm_issue(0);
  tdm_issue(1);

  const int NCHUNK = DIM / CHUNK_K;              // 16
  for (int c = 0; c < NCHUNK; ++c) {
    if (c < NCHUNK - 1) __builtin_amdgcn_s_wait_tensorcnt((short)1);
    else                __builtin_amdgcn_s_wait_tensorcnt((short)0);

    const float* bb = mybuf + (size_t)(c & 1) * BUF_DW + (size_t)ln * ROW_PITCH;
#pragma unroll
    for (int g = 0; g < 2; ++g) {                // two 32-K groups per chunk
      const int kb = c * CHUNK_K + g * 32;       // fragment-order offset == K base
      v16bf bhi, blo;
      split16(bb + g * 32 + half * 16, bhi, blo);   // B column fragment (16 K vals)

      v16bf ah0 = *(const v16bf*)(Ah0 + kb), al0 = *(const v16bf*)(Al0 + kb);
      v16bf ah1 = *(const v16bf*)(Ah1 + kb), al1 = *(const v16bf*)(Al1 + kb);
      v16bf ah2 = *(const v16bf*)(Ah2 + kb), al2 = *(const v16bf*)(Al2 + kb);
      v16bf ah3 = *(const v16bf*)(Ah3 + kb), al3 = *(const v16bf*)(Al3 + kb);

      // bf16x3: hi*hi + lo*hi + hi*lo  (f32 accumulate)
      c0 = __builtin_amdgcn_wmma_f32_16x16x32_bf16(false, ah0, false, bhi, (short)0, c0, false, false);
      c1 = __builtin_amdgcn_wmma_f32_16x16x32_bf16(false, ah1, false, bhi, (short)0, c1, false, false);
      c2 = __builtin_amdgcn_wmma_f32_16x16x32_bf16(false, ah2, false, bhi, (short)0, c2, false, false);
      c3 = __builtin_amdgcn_wmma_f32_16x16x32_bf16(false, ah3, false, bhi, (short)0, c3, false, false);
      c0 = __builtin_amdgcn_wmma_f32_16x16x32_bf16(false, al0, false, bhi, (short)0, c0, false, false);
      c1 = __builtin_amdgcn_wmma_f32_16x16x32_bf16(false, al1, false, bhi, (short)0, c1, false, false);
      c2 = __builtin_amdgcn_wmma_f32_16x16x32_bf16(false, al2, false, bhi, (short)0, c2, false, false);
      c3 = __builtin_amdgcn_wmma_f32_16x16x32_bf16(false, al3, false, bhi, (short)0, c3, false, false);
      c0 = __builtin_amdgcn_wmma_f32_16x16x32_bf16(false, ah0, false, blo, (short)0, c0, false, false);
      c1 = __builtin_amdgcn_wmma_f32_16x16x32_bf16(false, ah1, false, blo, (short)0, c1, false, false);
      c2 = __builtin_amdgcn_wmma_f32_16x16x32_bf16(false, ah2, false, blo, (short)0, c2, false, false);
      c3 = __builtin_amdgcn_wmma_f32_16x16x32_bf16(false, ah3, false, blo, (short)0, c3, false, false);
    }
    if (c + 2 < NCHUNK) {
      asm volatile("s_wait_dscnt 0x0" ::: "memory");   // LDS reads done before overwrite
      tdm_issue(c + 2);
    }
  }

  if (col < VOCAB) {
#pragma unroll
    for (int j = 0; j < 8; ++j) {
      const int m = j + 8 * half;
      logits[(size_t)(m     ) * VOCAB + col] = c0[j] * __builtin_amdgcn_rcpf(temps[m     ]);
      logits[(size_t)(m + 16) * VOCAB + col] = c1[j] * __builtin_amdgcn_rcpf(temps[m + 16]);
      logits[(size_t)(m + 32) * VOCAB + col] = c2[j] * __builtin_amdgcn_rcpf(temps[m + 32]);
      logits[(size_t)(m + 48) * VOCAB + col] = c3[j] * __builtin_amdgcn_rcpf(temps[m + 48]);
    }
  }
}

// ---------------------------------------------------------------------------
// Kernel 2: per-row softmax stats, top-k/top-p thresholds, sample, outputs.
// ---------------------------------------------------------------------------
__device__ __forceinline__ float hash_u01(unsigned long long x) {
  x ^= x >> 33; x *= 0xff51afd7ed558ccdULL;
  x ^= x >> 33; x *= 0xc4ceb9fe1a85ec53ULL;
  x ^= x >> 33;
  return (float)(x >> 40) * (1.0f / 16777216.0f);
}

__global__ __launch_bounds__(512) void sample_rows(
    const float* __restrict__ top_ps, const int* __restrict__ top_ks,
    float* __restrict__ out)
{
  const int row = blockIdx.x;
  const int t   = threadIdx.x;
  float* __restrict__ lrow = out + 2 * BATCH + (size_t)row * VOCAB;

  __shared__ float red[512];
  __shared__ float sh[4];

  float m = -INFINITY;
  for (int i = t; i < VOCAB; i += 512) m = fmaxf(m, lrow[i]);
  red[t] = m; __syncthreads();
  for (int s = 256; s > 0; s >>= 1) { if (t < s) red[t] = fmaxf(red[t], red[t + s]); __syncthreads(); }
  const float rmax = red[0]; __syncthreads();

  float acc = 0.0f;
  for (int i = t; i < VOCAB; i += 512) acc += __expf(lrow[i] - rmax);
  red[t] = acc; __syncthreads();
  for (int s = 256; s > 0; s >>= 1) { if (t < s) red[t] += red[t + s]; __syncthreads(); }
  const float rsum = red[0]; __syncthreads();
  const float inv  = 1.0f / rsum;
  const float pmax = inv;

  const float pp = top_ps[row];
  const float kk = (float)top_ks[row];

  float lo = 0.0f, hi = pmax;
  for (int it = 0; it < 20; ++it) {
    const float mid = 0.5f * (lo + hi);
    float cnt = 0.0f;
    for (int i = t; i < VOCAB; i += 512) {
      const float p = __expf(lrow[i] - rmax) * inv;
      cnt += (p > mid) ? 1.0f : 0.0f;
    }
    red[t] = cnt; __syncthreads();
    for (int s = 256; s > 0; s >>= 1) { if (t < s) red[t] += red[t + s]; __syncthreads(); }
    const float c0 = red[0]; __syncthreads();
    if (c0 > kk) lo = mid; else hi = mid;
  }
  const float tauk = lo;

  lo = 0.0f; hi = pmax;
  for (int it = 0; it < 20; ++it) {
    const float mid = 0.5f * (lo + hi);
    float mass = 0.0f;
    for (int i = t; i < VOCAB; i += 512) {
      const float p = __expf(lrow[i] - rmax) * inv;
      mass += (p > mid) ? p : 0.0f;
    }
    red[t] = mass; __syncthreads();
    for (int s = 256; s > 0; s >>= 1) { if (t < s) red[t] += red[t + s]; __syncthreads(); }
    const float m0 = red[0]; __syncthreads();
    if (m0 > pp) lo = mid; else hi = mid;
  }
  const float tau = fmaxf(tauk, lo);

  const int C   = (VOCAB + 511) / 512;
  const int beg = t * C;
  const int end = (beg + C < VOCAB) ? beg + C : VOCAB;
  float csum = 0.0f;
  for (int i = beg; i < end; ++i) {
    const float p = __expf(lrow[i] - rmax) * inv;
    if (p > tau || p == pmax) csum += p;
  }
  red[t] = csum; __syncthreads();
  if (t == 0) {
    float run = 0.0f;
    for (int i = 0; i < 512; ++i) { const float v = red[i]; red[i] = run; run += v; }
    sh[0] = run;
    out[row] = 0.0f;
    out[BATCH + row] = (lrow[0] - rmax) - __logf(rsum);
  }
  __syncthreads();
  const float total = sh[0];
  const float r = hash_u01(((unsigned long long)42u << 32) | (unsigned)row) * total;

  if (csum > 0.0f && r >= red[t] && r < red[t] + csum) {
    float a2 = red[t];
    int tok = beg;
    for (int i = beg; i < end; ++i) {
      const float p = __expf(lrow[i] - rmax) * inv;
      if (p > tau || p == pmax) {
        a2 += p;
        if (a2 > r) { tok = i; break; }
      }
    }
    out[row]         = (float)tok;
    out[BATCH + row] = (lrow[tok] - rmax) - __logf(rsum);
  }
  __syncthreads();

  for (int i = t; i < VOCAB; i += 512) {
    const float p = __expf(lrow[i] - rmax) * inv;
    lrow[i] = (p > tau || p == pmax) ? p : 0.0f;
  }
}

extern "C" void kernel_launch(void* const* d_in, const int* in_sizes, int n_in,
                              void* d_out, int out_size, void* d_ws, size_t ws_size,
                              hipStream_t stream) {
  const float* H      = (const float*)d_in[0];
  const float* E      = (const float*)d_in[1];
  const float* temps  = (const float*)d_in[2];
  const float* top_ps = (const float*)d_in[3];
  const int*   top_ks = (const int*)  d_in[4];
  float* out = (float*)d_out;

  __bf16* Ahi = (__bf16*)d_ws;                       // 64*1024 bf16 = 128 KB
  __bf16* Alo = Ahi + (size_t)BATCH * DIM;           // +128 KB

  float* logits = out + 2 * BATCH;
  const int strips = (VOCAB + 15) / 16;              // 3142
  const int blocks = (strips + NWAVE - 1) / NWAVE;   // 786

  split_hidden<<<(BATCH * DIM + 255) / 256, 256, 0, stream>>>(H, Ahi, Alo);
  lmhead_gemm_wmma<<<blocks, 128, 0, stream>>>(Ahi, Alo, E, temps, logits);
  sample_rows<<<BATCH, 512, 0, stream>>>(top_ps, top_ks, out);
}
